// TFNOBlock2D_24988119728189
// MI455X (gfx1250) — compile-verified, matrix operand order checked by reading, not run on hardware
//
#include <hip/hip_runtime.h>

#ifndef __has_builtin
#define __has_builtin(x) 0
#endif

#if defined(__HIP_DEVICE_COMPILE__)
#if __has_builtin(__builtin_amdgcn_global_load_async_to_lds_b32) && \
    __has_builtin(__builtin_amdgcn_s_wait_asynccnt)
#define HAVE_ASYNC_LDS 1
#endif
#endif

// ---------------------------------------------------------------------------
// Types / WMMA helpers (CDNA5 / gfx1250, wave32)
// ---------------------------------------------------------------------------
typedef __bf16 bf16_t;
typedef __attribute__((ext_vector_type(16))) __bf16 v16bf;
typedef __attribute__((ext_vector_type(8)))  float  v8f;

#if defined(HAVE_ASYNC_LDS)
typedef __attribute__((address_space(1))) int gint_t;   // global (device) int
typedef __attribute__((address_space(3))) int lint_t;   // LDS (shared) int
#endif

__device__ __forceinline__ bf16_t f2bf(float f) { return (bf16_t)f; }

__device__ __forceinline__ v8f wmma_bf(v16bf a, v16bf b, v8f c) {
  // D = A(16x32 bf16) * B(32x16 bf16) + C(16x16 f32)
  return __builtin_amdgcn_wmma_f32_16x16x32_bf16(
      false, a, false, b, (short)0, c, false, false);
}

__device__ __forceinline__ float gelu_f(float x) {
  return 0.5f * x * (1.0f + erff(x * 0.70710678118654752f));
}

// A-tile loader: 16 rows x 32 k, row-major source, ld in elements.
// ISA layout (16-bit A 16x32): lanes 0-15 -> rows 0-15, K {0..7,16..23};
// lanes 16-31 -> rows 0-15, K {8..15,24..31}. Works for B via Bt (n-major).
__device__ __forceinline__ v16bf load_tile16x32(const bf16_t* base, int ld) {
  int lane = threadIdx.x & 31;
  int r    = lane & 15;
  int kb   = (lane & 16) ? 8 : 0;
  const bf16_t* p = base + (size_t)r * ld + kb;
  v16bf v;
#pragma unroll
  for (int j = 0; j < 8; ++j) v[j] = p[j];
#pragma unroll
  for (int j = 0; j < 8; ++j) v[8 + j] = p[16 + j];
  return v;
}

// Same, but only 8 valid rows (rows 8..15 alias rows 0..7; results discarded).
__device__ __forceinline__ v16bf load_tile8x32(const bf16_t* base, int ld) {
  int lane = threadIdx.x & 31;
  int r    = lane & 7;
  int kb   = (lane & 16) ? 8 : 0;
  const bf16_t* p = base + (size_t)r * ld + kb;
  v16bf v;
#pragma unroll
  for (int j = 0; j < 8; ++j) v[j] = p[j];
#pragma unroll
  for (int j = 0; j < 8; ++j) v[8 + j] = p[16 + j];
  return v;
}

// ---------------------------------------------------------------------------
// Problem constants
// ---------------------------------------------------------------------------
#define BB   8
#define CC   256
#define HH   128            // NLAT (k)
#define WW   256            // NLON (w)
#define HWXY (HH * WW)      // 32768
#define NPOS (BB * HWXY)    // 262144 positions
#define BCK  (BB * CC * HH) // 262144 rows for DFT GEMM
#define MH   64             // kept l modes
#define MW   64             // kept m modes
#define HID  128
#define MP8  9              // LDS pad for 8 m-slots (fp32, conflict-free)

#define PI_F 3.14159265358979323846f

// ---------------------------------------------------------------------------
// Setup kernels: quadrature weights, DFT matrices, Legendre basis, w->bf16
// ---------------------------------------------------------------------------
__global__ void k_quad(float* wq) {
  int k = threadIdx.x;            // 0..127
  const int n = HH - 1;           // 127
  float s = 0.f;
  for (int jj = 1; jj <= n / 2; ++jj) {
    s += cosf(2.f * PI_F * k * jj / n) * (2.f / (4.f * jj * jj - 1.f));
  }
  float c = (k == 0 || k == n) ? 1.f : 2.f;
  wq[k] = (c / n) * (1.f - s);
}

__global__ void k_dft(bf16_t* Fr, bf16_t* Fi, bf16_t* Gt) {
  int m = blockIdx.x;             // 0..63
  int w = threadIdx.x;            // 0..255
  float ang = 2.f * PI_F * (float)m * (float)w / (float)WW;
  float sc  = 2.f * PI_F / (float)WW;
  Fr[m * WW + w] = f2bf(sc * cosf(ang));
  Fi[m * WW + w] = f2bf(-sc * sinf(ang));
  float sm = (m == 0) ? 1.f : 2.f;
  Gt[w * (2 * MW) + 2 * m]     = f2bf(sm * cosf(ang));
  Gt[w * (2 * MW) + 2 * m + 1] = f2bf(-sm * sinf(ang));
}

// Orthonormal associated Legendre, per-(m,k) thread, recurrence over l.
// Pf[m][l][k] = P * wq[k] (analysis), Ps[m][k][l] = P (synthesis, k-major).
__global__ void k_legendre(const float* __restrict__ wq, bf16_t* Pf, bf16_t* Ps) {
  int m = blockIdx.x;             // 0..63
  int k = threadIdx.x;            // 0..127
  float xk   = cosf(PI_F * (float)k / (float)(HH - 1));
  float sint = sqrtf(fmaxf(1.f - xk * xk, 0.f));
  float pmm  = sqrtf(1.f / (4.f * PI_F));
  for (int j = 0; j < m; ++j)
    pmm *= sqrtf((2.f * j + 3.f) / (2.f * j + 2.f)) * sint;
  float wk = wq[k];
  float p0 = 0.f, p1 = 0.f;
  for (int l = 0; l < MH; ++l) {
    float val;
    if (l < m)            val = 0.f;
    else if (l == m)      val = pmm;
    else if (l == m + 1)  val = sqrtf(2.f * m + 3.f) * xk * pmm;
    else {
      float lf = (float)l;
      float a  = sqrtf((4.f * lf * lf - 1.f) / (lf * lf - (float)(m * m)));
      float bc = sqrtf(((lf - 1.f) * (lf - 1.f) - (float)(m * m)) /
                       (4.f * (lf - 1.f) * (lf - 1.f) - 1.f));
      val = a * (xk * p1 - bc * p0);
    }
    p0 = p1; p1 = val;
    Pf[((size_t)m * MH + l) * HH + k] = f2bf(val * wk);
    Ps[((size_t)m * HH + k) * MH + l] = f2bf(val);
  }
}

__global__ void k_cvtw(const float* sw, const float* f1, const float* f2,
                       bf16_t* swb, bf16_t* f1b, bf16_t* f2b) {
  int i = blockIdx.x * 256 + threadIdx.x;
  if (i < CC * CC) swb[i] = f2bf(sw[i]);
  if (i < HID * CC) { f1b[i] = f2bf(f1[i]); f2b[i] = f2bf(f2[i]); }
}

// ---------------------------------------------------------------------------
// LayerNorm over channel dim. Block = 32 positions x 256 channels.
// Writes normalized y in (b,c,hw) bf16 and raw x transposed (b,hw,c) bf16.
// ---------------------------------------------------------------------------
__global__ void k_ln1(const float* __restrict__ x, const float* __restrict__ g,
                      const float* __restrict__ be, bf16_t* y, bf16_t* xT) {
  int pos0 = blockIdx.x * 32;
  int b    = pos0 / HWXY;
  int hw   = (pos0 % HWXY) + (threadIdx.x & 31);
  int grp  = threadIdx.x >> 5;            // 0..7, channels grp*32..+31
  int c0   = grp * 32;
  const float* xb = x + (size_t)b * CC * HWXY + hw;
  float vals[32];
  float s = 0.f, s2 = 0.f;
#pragma unroll
  for (int j = 0; j < 32; ++j) {
    float v = xb[(size_t)(c0 + j) * HWXY];
    vals[j] = v; s += v; s2 += v * v;
  }
  __shared__ float red[2][8][32];
  int wi = threadIdx.x & 31;
  red[0][grp][wi] = s; red[1][grp][wi] = s2;
  __syncthreads();
  float ts = 0.f, ts2 = 0.f;
#pragma unroll
  for (int q = 0; q < 8; ++q) { ts += red[0][q][wi]; ts2 += red[1][q][wi]; }
  float mu = ts / (float)CC;
  float rstd = rsqrtf(ts2 / (float)CC - mu * mu + 1e-5f);
  bf16_t* yb = y + (size_t)b * CC * HWXY + hw;
  bf16_t* xt = xT + ((size_t)b * HWXY + hw) * CC + c0;
#pragma unroll
  for (int j = 0; j < 32; ++j) {
    int c = c0 + j; float v = vals[j];
    yb[(size_t)c * HWXY] = f2bf((v - mu) * rstd * g[c] + be[c]);
    xt[j] = f2bf(v);
  }
}

// LN2: input (b,c,hw) f32, output transposed (b,hw,c) bf16 only.
__global__ void k_ln2(const float* __restrict__ s, const float* __restrict__ g,
                      const float* __restrict__ be, bf16_t* zT) {
  int pos0 = blockIdx.x * 32;
  int b    = pos0 / HWXY;
  int hw   = (pos0 % HWXY) + (threadIdx.x & 31);
  int grp  = threadIdx.x >> 5;
  int c0   = grp * 32;
  const float* xb = s + (size_t)b * CC * HWXY + hw;
  float vals[32];
  float su = 0.f, s2 = 0.f;
#pragma unroll
  for (int j = 0; j < 32; ++j) {
    float v = xb[(size_t)(c0 + j) * HWXY];
    vals[j] = v; su += v; s2 += v * v;
  }
  __shared__ float red[2][8][32];
  int wi = threadIdx.x & 31;
  red[0][grp][wi] = su; red[1][grp][wi] = s2;
  __syncthreads();
  float ts = 0.f, ts2 = 0.f;
#pragma unroll
  for (int q = 0; q < 8; ++q) { ts += red[0][q][wi]; ts2 += red[1][q][wi]; }
  float mu = ts / (float)CC;
  float rstd = rsqrtf(ts2 / (float)CC - mu * mu + 1e-5f);
  bf16_t* zt = zT + ((size_t)b * HWXY + hw) * CC + c0;
#pragma unroll
  for (int j = 0; j < 32; ++j) {
    int c = c0 + j;
    zt[j] = f2bf((vals[j] - mu) * rstd * g[c] + be[c]);
  }
}

// ---------------------------------------------------------------------------
// Forward DFT GEMM: xf[m][bck] = sum_w y[bck][w] * F[m][w]   (re & im)
// M = 262144 rows (b,c,k), N = 64 (m), K = 256 (w).
// Grid (rows/128, 2): each block stages 32 m-rows of Fr/Fi in LDS (32 KB).
// ---------------------------------------------------------------------------
__global__ void k_fwd_dft(const bf16_t* __restrict__ y, const bf16_t* __restrict__ Fr,
                          const bf16_t* __restrict__ Fi, bf16_t* xf_re, bf16_t* xf_im) {
  __shared__ bf16_t sFr[32 * WW];
  __shared__ bf16_t sFi[32 * WW];
  int mBase = blockIdx.y * 32;
#pragma unroll 4
  for (int rep = 0; rep < 32; ++rep) {
    int idx = threadIdx.x + 256 * rep;       // < 8192
    sFr[idx] = Fr[(size_t)mBase * WW + idx];
    sFi[idx] = Fi[(size_t)mBase * WW + idx];
  }
  __syncthreads();
  int wv = threadIdx.x >> 5;
  int rowBase = blockIdx.x * 128 + wv * 16;
  v8f z = {};
  v8f accR[2], accI[2];
#pragma unroll
  for (int nt = 0; nt < 2; ++nt) { accR[nt] = z; accI[nt] = z; }
#pragma unroll
  for (int ks = 0; ks < 8; ++ks) {
    v16bf a = load_tile16x32(y + (size_t)rowBase * WW + ks * 32, WW);
#pragma unroll
    for (int nt = 0; nt < 2; ++nt) {
      v16bf br = load_tile16x32(sFr + (nt * 16) * WW + ks * 32, WW);
      v16bf bi = load_tile16x32(sFi + (nt * 16) * WW + ks * 32, WW);
      accR[nt] = wmma_bf(a, br, accR[nt]);
      accI[nt] = wmma_bf(a, bi, accI[nt]);
    }
  }
  int lane = threadIdx.x & 31;
  int cn = lane & 15, mb = (lane & 16) ? 8 : 0;
#pragma unroll
  for (int nt = 0; nt < 2; ++nt) {
    size_t base = (size_t)(mBase + nt * 16 + cn) * BCK;
#pragma unroll
    for (int v = 0; v < 8; ++v) {
      int r = rowBase + mb + v;
      xf_re[base + r] = f2bf(accR[nt][v]);
      xf_im[base + r] = f2bf(accI[nt][v]);
    }
  }
}

// ---------------------------------------------------------------------------
// Forward Legendre, per m: xl[bc][l] = sum_k xf[bc][k] * Pf[m][l][k]
// M = 2048 (b,c), N = 64 (l), K = 128 (k). Pf[m] staged in LDS (16 KB).
// Output scattered to [l][m][b][c].
// ---------------------------------------------------------------------------
__global__ void k_fwd_leg(const bf16_t* __restrict__ xf_re, const bf16_t* __restrict__ xf_im,
                          const bf16_t* __restrict__ Pf, bf16_t* xl_re, bf16_t* xl_im) {
  __shared__ bf16_t sPf[MH * HH];            // 64 x 128
  int m = blockIdx.y;
#pragma unroll 4
  for (int rep = 0; rep < 32; ++rep) {
    int idx = threadIdx.x + 256 * rep;       // < 8192
    sPf[idx] = Pf[(size_t)m * MH * HH + idx];
  }
  __syncthreads();
  int wv = threadIdx.x >> 5;
  int rowBase = blockIdx.x * 128 + wv * 16;
  const bf16_t* Ar = xf_re + (size_t)m * BCK + (size_t)rowBase * HH;
  const bf16_t* Ai = xf_im + (size_t)m * BCK + (size_t)rowBase * HH;
  v8f z = {};
  v8f accR[4], accI[4];
#pragma unroll
  for (int nt = 0; nt < 4; ++nt) { accR[nt] = z; accI[nt] = z; }
#pragma unroll
  for (int ks = 0; ks < 4; ++ks) {
    v16bf ar = load_tile16x32(Ar + ks * 32, HH);
    v16bf ai = load_tile16x32(Ai + ks * 32, HH);
#pragma unroll
    for (int nt = 0; nt < 4; ++nt) {
      v16bf bt = load_tile16x32(sPf + (nt * 16) * HH + ks * 32, HH);
      accR[nt] = wmma_bf(ar, bt, accR[nt]);
      accI[nt] = wmma_bf(ai, bt, accI[nt]);
    }
  }
  int lane = threadIdx.x & 31;
  int cn = lane & 15, mb = (lane & 16) ? 8 : 0;
#pragma unroll
  for (int nt = 0; nt < 4; ++nt) {
    int lcol = nt * 16 + cn;
#pragma unroll
    for (int v = 0; v < 8; ++v) {
      int r = rowBase + mb + v;
      size_t a_ = (size_t)(lcol * MW + m) * 2048 + r;
      xl_re[a_] = f2bf(accR[nt][v]);
      xl_im[a_] = f2bf(accI[nt][v]);
    }
  }
}

// ---------------------------------------------------------------------------
// Spectral conv: per (l,m): yf[b][o] = sum_i xl[b][i] * (wr + i wi)[o][i][l][m]
// Grid (l=64, o-tile=16, m-block=8x8). Weights streamed ONCE, staged in LDS
// as fp32 via gfx1250 async global->LDS copies (ASYNCcnt) when available;
// bf16 conversion happens while building WMMA B-tiles.
// ---------------------------------------------------------------------------
__global__ void k_spectral(const bf16_t* __restrict__ xl_re, const bf16_t* __restrict__ xl_im,
                           const float* __restrict__ wr, const float* __restrict__ wi_,
                           bf16_t* yf_re, bf16_t* yf_im) {
  __shared__ float lds_r[16 * 32 * MP8];     // [o][i][m(8) pad 9] fp32
  __shared__ float lds_i[16 * 32 * MP8];
  int l  = blockIdx.x;
  int o0 = blockIdx.y * 16;
  int m0 = blockIdx.z * 8;
  int tid = threadIdx.x;
  int wv  = tid >> 5;
  int m   = m0 + wv;                         // one m per wave
  v8f a_rr = {}, a_ii = {}, a_im = {};

  for (int ib = 0; ib < 8; ++ib) {
    int i0 = ib * 32;
    __syncthreads();                         // previous tile fully consumed
    // cooperative stage: 16o x 32i x 8m = 4096 fp32 per array (16 reps)
#pragma unroll 4
    for (int rep = 0; rep < 16; ++rep) {
      int idx = tid + 256 * rep;
      int mm = idx & 7;
      int ii = (idx >> 3) & 31;
      int oo = idx >> 8;
      size_t gi = (((size_t)(o0 + oo) * CC + (i0 + ii)) * MH + l) * MW + (m0 + mm);
      int loff = (oo * 32 + ii) * MP8 + mm;
#if defined(HAVE_ASYNC_LDS)
      __builtin_amdgcn_global_load_async_to_lds_b32(
          (gint_t*)(wr + gi), (lint_t*)&lds_r[loff], 0, 0);
      __builtin_amdgcn_global_load_async_to_lds_b32(
          (gint_t*)(wi_ + gi), (lint_t*)&lds_i[loff], 0, 0);
#else
      lds_r[loff] = wr[gi];
      lds_i[loff] = wi_[gi];
      if (ib < 7 && rep == 0)
        __builtin_prefetch(&wr[gi + 32 * (size_t)(MH * MW)], 0, 1);
#endif
    }
#if defined(HAVE_ASYNC_LDS)
    __builtin_amdgcn_s_wait_asynccnt(0);
#endif
    __syncthreads();
    // A tiles (8 valid rows = batch), bf16 from global
    v16bf ar = load_tile8x32(xl_re + (size_t)(l * MW + m) * 2048 + i0, CC);
    v16bf ai = load_tile8x32(xl_im + (size_t)(l * MW + m) * 2048 + i0, CC);
    // B tiles from LDS (Bt[n=o][k=i] at this wave's m slot), cvt fp32->bf16
    int lane = tid & 31;
    int on = lane & 15;
    int kb = (lane & 16) ? 8 : 0;
    v16bf br, bi;
#pragma unroll
    for (int j = 0; j < 8; ++j) {
      br[j]     = f2bf(lds_r[(on * 32 + kb + j) * MP8 + wv]);
      br[8 + j] = f2bf(lds_r[(on * 32 + 16 + kb + j) * MP8 + wv]);
      bi[j]     = f2bf(lds_i[(on * 32 + kb + j) * MP8 + wv]);
      bi[8 + j] = f2bf(lds_i[(on * 32 + 16 + kb + j) * MP8 + wv]);
    }
    a_rr = wmma_bf(ar, br, a_rr);
    a_ii = wmma_bf(ai, bi, a_ii);
    a_im = wmma_bf(ar, bi, a_im);
    a_im = wmma_bf(ai, br, a_im);
  }
  // store: rows b = 0..7 live in lanes 0..15, VGPRs 0..7
  int lane = tid & 31;
  if (lane < 16) {
#pragma unroll
    for (int v = 0; v < 8; ++v) {
      size_t off = (((size_t)m * BB + v) * CC + (o0 + lane)) * MH + l;
      yf_re[off] = f2bf(a_rr[v] - a_ii[v]);
      yf_im[off] = f2bf(a_im[v]);
    }
  }
}

// ---------------------------------------------------------------------------
// Inverse Legendre, per m: yk[bo][k] = sum_l yf[bo][l] * Ps[m][k][l]
// M = 2048 (b,o), N = 128 (k), K = 64 (l). Ps[m] staged in LDS (16 KB).
// Output interleaved yk[bok][2m|2m+1].
// ---------------------------------------------------------------------------
__global__ void k_inv_leg(const bf16_t* __restrict__ yf_re, const bf16_t* __restrict__ yf_im,
                          const bf16_t* __restrict__ Ps, bf16_t* yk) {
  __shared__ bf16_t sPs[HH * MH];            // 128 x 64
  int m = blockIdx.y;
#pragma unroll 4
  for (int rep = 0; rep < 32; ++rep) {
    int idx = threadIdx.x + 256 * rep;       // < 8192
    sPs[idx] = Ps[(size_t)m * HH * MH + idx];
  }
  __syncthreads();
  int wv = threadIdx.x >> 5;
  int rowBase = blockIdx.x * 32 + (wv & 1) * 16;
  int ntBase  = (wv >> 1) * 2;
  const bf16_t* Ar = yf_re + (size_t)m * 2048 * MH + (size_t)rowBase * MH;
  const bf16_t* Ai = yf_im + (size_t)m * 2048 * MH + (size_t)rowBase * MH;
  v8f z = {};
  v8f cR[2], cI[2];
#pragma unroll
  for (int q = 0; q < 2; ++q) { cR[q] = z; cI[q] = z; }
#pragma unroll
  for (int ks = 0; ks < 2; ++ks) {
    v16bf ar = load_tile16x32(Ar + ks * 32, MH);
    v16bf ai = load_tile16x32(Ai + ks * 32, MH);
#pragma unroll
    for (int q = 0; q < 2; ++q) {
      int nt = ntBase + q;
      v16bf bt = load_tile16x32(sPs + (nt * 16) * MH + ks * 32, MH);
      cR[q] = wmma_bf(ar, bt, cR[q]);
      cI[q] = wmma_bf(ai, bt, cI[q]);
    }
  }
  int lane = threadIdx.x & 31;
  int cn = lane & 15, mb = (lane & 16) ? 8 : 0;
#pragma unroll
  for (int q = 0; q < 2; ++q) {
    int kcol = (ntBase + q) * 16 + cn;
#pragma unroll
    for (int v = 0; v < 8; ++v) {
      int r = rowBase + mb + v;
      size_t a_ = ((size_t)r * HH + kcol) * (2 * MW) + 2 * m;
      yk[a_]     = f2bf(cR[q][v]);
      yk[a_ + 1] = f2bf(cI[q][v]);
    }
  }
}

// ---------------------------------------------------------------------------
// Inverse DFT GEMM: s[bok][w] = sum_m (re*Gc - im*Gs); A = yk (K=128), B = Gt.
// Grid (rows/32, 2): each block stages 128 w-rows of Gt in LDS (32 KB).
// ---------------------------------------------------------------------------
__global__ void k_inv_dft(const bf16_t* __restrict__ yk, const bf16_t* __restrict__ Gt,
                          float* __restrict__ sbuf) {
  __shared__ bf16_t sGt[128 * 128];
  int wBase = blockIdx.y * 128;
#pragma unroll 4
  for (int rep = 0; rep < 64; ++rep) {
    int idx = threadIdx.x + 256 * rep;       // < 16384
    sGt[idx] = Gt[(size_t)wBase * (2 * MW) + idx];
  }
  __syncthreads();
  int wv = threadIdx.x >> 5;
  int rowBase = blockIdx.x * 32 + (wv & 1) * 16;
  int ntBase  = (wv >> 1) * 2;               // local n-tiles, 8 total
  v8f z = {};
  v8f c[2]; c[0] = z; c[1] = z;
#pragma unroll
  for (int ks = 0; ks < 4; ++ks) {
    v16bf a = load_tile16x32(yk + (size_t)rowBase * (2 * MW) + ks * 32, 2 * MW);
#pragma unroll
    for (int q = 0; q < 2; ++q) {
      int nt = ntBase + q;
      v16bf bt = load_tile16x32(sGt + (nt * 16) * (2 * MW) + ks * 32, 2 * MW);
      c[q] = wmma_bf(a, bt, c[q]);
    }
  }
  int lane = threadIdx.x & 31;
  int cn = lane & 15, mb = (lane & 16) ? 8 : 0;
#pragma unroll
  for (int q = 0; q < 2; ++q) {
    int wcol = wBase + (ntBase + q) * 16 + cn;
#pragma unroll
    for (int v = 0; v < 8; ++v)
      sbuf[(size_t)(rowBase + mb + v) * WW + wcol] = c[q][v];
  }
}

// ---------------------------------------------------------------------------
// Skip 1x1 conv: out[b][o][hw] = sum_i x[b][i][hw]*W[o][i] + bias[o]
// A = skip_w (o x i) staged in LDS (32 KB), Bt = xT (p x i). 64 o x 64 p.
// ---------------------------------------------------------------------------
__global__ void k_skip(const bf16_t* __restrict__ xT, const bf16_t* __restrict__ swb,
                       const float* __restrict__ sb, float* __restrict__ skbuf) {
  __shared__ bf16_t sW[64 * CC];
  int p0 = blockIdx.x * 64;
  int o0 = blockIdx.y * 64;
#pragma unroll 4
  for (int rep = 0; rep < 64; ++rep) {
    int idx = threadIdx.x + 256 * rep;       // < 16384
    sW[idx] = swb[(size_t)o0 * CC + idx];
  }
  __syncthreads();
  int wv = threadIdx.x >> 5;
  int mt = wv & 3;           // o sub-tile
  int ng = wv >> 2;          // 0..1, two p-tiles each
  v8f z = {};
  v8f c[2]; c[0] = z; c[1] = z;
#pragma unroll
  for (int ks = 0; ks < 8; ++ks) {
    v16bf a = load_tile16x32(sW + (mt * 16) * CC + ks * 32, CC);
#pragma unroll
    for (int q = 0; q < 2; ++q) {
      int nt = ng * 2 + q;
      v16bf bt = load_tile16x32(xT + (size_t)(p0 + nt * 16) * CC + ks * 32, CC);
      c[q] = wmma_bf(a, bt, c[q]);
    }
  }
  int lane = threadIdx.x & 31;
  int cn = lane & 15, mb = (lane & 16) ? 8 : 0;
#pragma unroll
  for (int q = 0; q < 2; ++q) {
    int p = p0 + (ng * 2 + q) * 16 + cn;
    int b = p >> 15, hw = p & (HWXY - 1);
#pragma unroll
    for (int v = 0; v < 8; ++v) {
      int o = o0 + mt * 16 + mb + v;
      skbuf[((size_t)(b * CC + o)) * HWXY + hw] = c[q][v] + sb[o];
    }
  }
}

// g = gelu(isht + skip), in place over sbuf
__global__ void k_combine1(float* s, const float* __restrict__ sk) {
  size_t i0 = (size_t)blockIdx.x * 1024 + threadIdx.x;
#pragma unroll
  for (int j = 0; j < 4; ++j) {
    size_t i = i0 + (size_t)j * 256;
    s[i] = gelu_f(s[i] + sk[i]);
  }
}

// fc1: h1T[p][o'] = gelu( sum_i zT[p][i]*fc1_w[o'][i] + b[o'] )
__global__ void k_fc1(const bf16_t* __restrict__ zT, const bf16_t* __restrict__ f1b,
                      const float* __restrict__ b1, bf16_t* __restrict__ h1T) {
  int wv = threadIdx.x >> 5;
  int rowBase = blockIdx.x * 32 + (wv & 1) * 16;
  int ntg = wv >> 1;
  v8f z = {};
  v8f c[2]; c[0] = z; c[1] = z;
#pragma unroll
  for (int ks = 0; ks < 8; ++ks) {
    v16bf a = load_tile16x32(zT + (size_t)rowBase * CC + ks * 32, CC);
#pragma unroll
    for (int q = 0; q < 2; ++q) {
      int nt = ntg * 2 + q;
      v16bf bt = load_tile16x32(f1b + (size_t)(nt * 16) * CC + ks * 32, CC);
      c[q] = wmma_bf(a, bt, c[q]);
    }
  }
  int lane = threadIdx.x & 31;
  int cn = lane & 15, mb = (lane & 16) ? 8 : 0;
#pragma unroll
  for (int q = 0; q < 2; ++q) {
    int o = (ntg * 2 + q) * 16 + cn;
#pragma unroll
    for (int v = 0; v < 8; ++v) {
      int p = rowBase + mb + v;
      h1T[(size_t)p * HID + o] = f2bf(gelu_f(c[q][v] + b1[o]));
    }
  }
}

// fc2: z2[p][o] = sum_h h1T[p][h]*fc2_w[o][h] + b[o]
__global__ void k_fc2(const bf16_t* __restrict__ h1T, const bf16_t* __restrict__ f2b,
                      const float* __restrict__ b2, float* __restrict__ z2) {
  int wv = threadIdx.x >> 5;
  int rowBase = blockIdx.x * 16;
  v8f z = {};
  v8f c[2]; c[0] = z; c[1] = z;
#pragma unroll
  for (int ks = 0; ks < 4; ++ks) {
    v16bf a = load_tile16x32(h1T + (size_t)rowBase * HID + ks * 32, HID);
#pragma unroll
    for (int q = 0; q < 2; ++q) {
      int nt = wv * 2 + q;
      v16bf bt = load_tile16x32(f2b + (size_t)(nt * 16) * HID + ks * 32, HID);
      c[q] = wmma_bf(a, bt, c[q]);
    }
  }
  int lane = threadIdx.x & 31;
  int cn = lane & 15, mb = (lane & 16) ? 8 : 0;
#pragma unroll
  for (int q = 0; q < 2; ++q) {
    int o = (wv * 2 + q) * 16 + cn;
#pragma unroll
    for (int v = 0; v < 8; ++v) {
      int p = rowBase + mb + v;
      z2[(size_t)p * CC + o] = c[q][v] + b2[o];
    }
  }
}

// out = gelu(g + z2), g in (b,o,hw), z2 in (b,hw,o)
__global__ void k_combine2(const float* __restrict__ g, const float* __restrict__ z2,
                           float* __restrict__ out) {
  size_t i = (size_t)blockIdx.x * 256 + threadIdx.x;
  int b  = (int)(i / ((size_t)CC * HWXY));
  int r  = (int)(i % ((size_t)CC * HWXY));
  int o  = r / HWXY;
  int hw = r % HWXY;
  float zz = z2[((size_t)b * HWXY + hw) * CC + o];
  out[i] = gelu_f(g[i] + zz);
}

// ---------------------------------------------------------------------------
// Host: workspace layout + launch sequence
// ---------------------------------------------------------------------------
extern "C" void kernel_launch(void* const* d_in, const int* in_sizes, int n_in,
                              void* d_out, int out_size, void* d_ws, size_t ws_size,
                              hipStream_t stream) {
  const float* x      = (const float*)d_in[0];
  const float* ln1_g  = (const float*)d_in[1];
  const float* ln1_b  = (const float*)d_in[2];
  const float* wsr    = (const float*)d_in[3];
  const float* wsi    = (const float*)d_in[4];
  const float* skip_w = (const float*)d_in[5];
  const float* skip_b = (const float*)d_in[6];
  const float* ln2_g  = (const float*)d_in[7];
  const float* ln2_b  = (const float*)d_in[8];
  const float* fc1_w  = (const float*)d_in[9];
  const float* fc1_b  = (const float*)d_in[10];
  const float* fc2_w  = (const float*)d_in[11];
  const float* fc2_b  = (const float*)d_in[12];
  float* out = (float*)d_out;

  char* ws = (char*)d_ws;
  size_t off = 0;
  auto alloc = [&](size_t bytes) -> char* {
    char* p = ws + off;
    off += (bytes + 255) & ~(size_t)255;
    return p;
  };
  float*  wq    = (float*)alloc(HH * 4);
  bf16_t* Fr    = (bf16_t*)alloc((size_t)MW * WW * 2);
  bf16_t* Fi    = (bf16_t*)alloc((size_t)MW * WW * 2);
  bf16_t* Gt    = (bf16_t*)alloc((size_t)WW * 2 * MW * 2);
  bf16_t* Pf    = (bf16_t*)alloc((size_t)MW * MH * HH * 2);
  bf16_t* Ps    = (bf16_t*)alloc((size_t)MW * HH * MH * 2);
  bf16_t* swb   = (bf16_t*)alloc((size_t)CC * CC * 2);
  bf16_t* f1b   = (bf16_t*)alloc((size_t)HID * CC * 2);
  bf16_t* f2b   = (bf16_t*)alloc((size_t)CC * HID * 2);
  bf16_t* yb    = (bf16_t*)alloc((size_t)BB * CC * HWXY * 2);       // LN1(x), (b,c,hw)
  bf16_t* xT    = (bf16_t*)alloc((size_t)NPOS * CC * 2);            // x^T; reused as zT
  bf16_t* xf_re = (bf16_t*)alloc((size_t)MW * BCK * 2);             // reused as h1T (with xf_im)
  bf16_t* xf_im = (bf16_t*)alloc((size_t)MW * BCK * 2);
  bf16_t* xl_re = (bf16_t*)alloc((size_t)MH * MW * 2048 * 2);
  bf16_t* xl_im = (bf16_t*)alloc((size_t)MH * MW * 2048 * 2);
  bf16_t* yf_re = (bf16_t*)alloc((size_t)MW * 2048 * MH * 2);
  bf16_t* yf_im = (bf16_t*)alloc((size_t)MW * 2048 * MH * 2);
  bf16_t* yk    = (bf16_t*)alloc((size_t)BCK * 2 * MW * 2);
  float*  sbuf  = (float*)alloc((size_t)BB * CC * HWXY * 4);
  float*  skbuf = (float*)alloc((size_t)BB * CC * HWXY * 4);        // reused as z2
  (void)ws_size; (void)in_sizes; (void)n_in; (void)out_size;

  bf16_t* zT  = xT;            // LN2 output reuses x^T
  bf16_t* h1T = xf_re;         // fc1 output reuses xf_re||xf_im (67 MB contiguous)
  float*  z2  = skbuf;         // fc2 output reuses skip buffer

  // constants
  k_quad<<<1, 128, 0, stream>>>(wq);
  k_dft<<<64, 256, 0, stream>>>(Fr, Fi, Gt);
  k_legendre<<<64, 128, 0, stream>>>(wq, Pf, Ps);
  k_cvtw<<<256, 256, 0, stream>>>(skip_w, fc1_w, fc2_w, swb, f1b, f2b);

  // pipeline
  k_ln1<<<NPOS / 32, 256, 0, stream>>>(x, ln1_g, ln1_b, yb, xT);
  k_fwd_dft<<<dim3(BCK / 128, 2), 256, 0, stream>>>(yb, Fr, Fi, xf_re, xf_im);
  k_fwd_leg<<<dim3(16, 64), 256, 0, stream>>>(xf_re, xf_im, Pf, xl_re, xl_im);
  k_spectral<<<dim3(64, 16, 8), 256, 0, stream>>>(xl_re, xl_im, wsr, wsi, yf_re, yf_im);
  k_inv_leg<<<dim3(64, 64), 256, 0, stream>>>(yf_re, yf_im, Ps, yk);
  k_inv_dft<<<dim3(BCK / 32, 2), 256, 0, stream>>>(yk, Gt, sbuf);
  k_skip<<<dim3(NPOS / 64, 4), 256, 0, stream>>>(xT, swb, skip_b, skbuf);
  k_combine1<<<(BB * CC * HWXY) / 1024, 256, 0, stream>>>(sbuf, skbuf);
  k_ln2<<<NPOS / 32, 256, 0, stream>>>(sbuf, ln2_g, ln2_b, zT);
  k_fc1<<<NPOS / 32, 256, 0, stream>>>(zT, f1b, fc1_b, h1T);
  k_fc2<<<NPOS / 16, 256, 0, stream>>>(h1T, f2b, fc2_b, z2);
  k_combine2<<<(BB * CC * HWXY) / 256, 256, 0, stream>>>(sbuf, z2, out);
}
